// Mamba2Mixer_62869731278948
// MI455X (gfx1250) — matrix operand order repaired and checked by reading, not compiled, not run on hardware
//
#include <hip/hip_runtime.h>

// ---------------------------------------------------------------------------
// Mamba2 mixer on MI455X (gfx1250):
//  - all matmul FLOPs on v_wmma_f32_16x16x32_bf16
//  - global->LDS staging via global_load_async_to_lds_b128 (ASYNCcnt),
//    double-buffered K pipeline in the GEMMs
//  - TDM tensor_load_to_lds (TENSORcnt) for the prev-state tile
//  - all WMMA operands stored K-contiguous in LDS (ds_load_b128 fragments)
// ---------------------------------------------------------------------------

typedef __bf16 bf16;
typedef __attribute__((ext_vector_type(16))) __bf16        v16bf;
typedef __attribute__((ext_vector_type(8)))  float         v8f;
typedef __attribute__((ext_vector_type(4)))  int           v4i;
typedef __attribute__((ext_vector_type(8)))  int           v8i;
typedef __attribute__((ext_vector_type(4)))  unsigned int  v4u;

#define D_MODEL   2048
#define D_STATE   128
#define NHEADS    64
#define HEAD_DIM  64
#define D_INNER   4096            // NHEADS*HEAD_DIM
#define CONV_DIM  4352            // D_INNER + 2*D_STATE
#define D_IN_PROJ 8512            // 2*D_INNER + 2*D_STATE + NHEADS
#define CHUNK     128
#define NCHUNK    32              // SEQ / CHUNK
#define BATCH     2
#define SEQ       4096
#define RMS_EPS   1e-5f

// ---------------- async global->LDS (CDNA5, ASYNCcnt) -----------------------

#if (__has_builtin(__builtin_amdgcn_global_load_async_to_lds_b128) && \
     __has_builtin(__builtin_amdgcn_s_wait_asynccnt))
#define HAVE_ASYNC_LDS 1
#else
#define HAVE_ASYNC_LDS 0
#endif

#if HAVE_ASYNC_LDS
#define WAIT_ASYNC(n) __builtin_amdgcn_s_wait_asynccnt(n)
#else
#define WAIT_ASYNC(n)
#endif

// ---------------- TDM tensor DMA (CDNA5, TENSORcnt) -------------------------

#if (__has_builtin(__builtin_amdgcn_tensor_load_to_lds) && \
     __has_builtin(__builtin_amdgcn_s_wait_tensorcnt))
#define HAVE_TDM 1
#else
#define HAVE_TDM 0
#endif

__device__ __forceinline__ void async_ld_b128(void* lds, const void* g) {
#if HAVE_ASYNC_LDS
  __builtin_amdgcn_global_load_async_to_lds_b128(
      (__attribute__((address_space(1))) v4i*)g,
      (__attribute__((address_space(3))) v4i*)lds, 0, 0);
#else
  *reinterpret_cast<uint4*>(lds) = *reinterpret_cast<const uint4*>(g);
#endif
}

// ---------------- WMMA fragment helpers (ISA 7.12.2 layouts, wave32) --------

// A fragment 16x32 bf16 from row-major LDS (ld in elements).
// Per-lane data is two contiguous 16B runs -> paired ds_load_b128.
__device__ __forceinline__ v16bf load_frag_a(const bf16* src, int ld) {
  int lane = threadIdx.x & 31;
  int m = lane & 15, hh = lane >> 4;
  v16bf a;
#pragma unroll
  for (int i = 0; i < 16; ++i) {
    int k = ((i & 8) << 1) + hh * 8 + (i & 7);   // (i<8?0:16) + hh*8 + i%8
    a[i] = src[m * ld + k];
  }
  return a;
}

// B fragment where the logical B[k][n] lives transposed in LDS: src[n*ld + k].
// Per-lane data is two contiguous 16B runs -> paired ds_load_b128.
__device__ __forceinline__ v16bf load_frag_bt(const bf16* src, int ld) {
  int lane = threadIdx.x & 31;
  int n = lane & 15, hh = lane >> 4;
  v16bf b;
#pragma unroll
  for (int i = 0; i < 16; ++i) {
    int k = ((i & 8) << 1) + hh * 8 + (i & 7);
    b[i] = src[n * ld + k];
  }
  return b;
}

__device__ __forceinline__ v8f zero8() {
  v8f z = {0.f, 0.f, 0.f, 0.f, 0.f, 0.f, 0.f, 0.f};
  return z;
}

__device__ __forceinline__ float siluf(float x) { return x / (1.f + __expf(-x)); }

// ---------------- fp32 -> bf16 convert (plain + transposing) ----------------

__global__ void k_f32_to_bf16(const float* __restrict__ src,
                              bf16* __restrict__ dst, int n) {
  int i = (blockIdx.x * 256 + threadIdx.x) * 4;
  if (i + 3 < n) {
    float4 f = *reinterpret_cast<const float4*>(src + i);
    dst[i + 0] = (bf16)f.x;
    dst[i + 1] = (bf16)f.y;
    dst[i + 2] = (bf16)f.z;
    dst[i + 3] = (bf16)f.w;
  } else {
    for (; i < n; ++i) dst[i] = (bf16)src[i];
  }
}

// src: [R][C] fp32 row-major  ->  dst: [C][R] bf16 row-major (32x32 tiles)
__global__ __launch_bounds__(256) void k_f32_to_bf16_T(
    const float* __restrict__ src, bf16* __restrict__ dst, int R, int C) {
  __shared__ float t[32][33];
  int r0 = blockIdx.y * 32, c0 = blockIdx.x * 32;
  int lr = threadIdx.x & 31, lc = threadIdx.x >> 5;   // 8 rows per pass
#pragma unroll
  for (int i = 0; i < 32; i += 8)
    t[lc + i][lr] = src[(size_t)(r0 + lc + i) * C + c0 + lr];
  __syncthreads();
#pragma unroll
  for (int i = 0; i < 32; i += 8)
    dst[(size_t)(c0 + lc + i) * R + r0 + lr] = (bf16)t[lr][lc + i];
}

// ---------------- generic bf16 WMMA GEMM, C = A @ BT^T ----------------------
// A: [M][K] bf16, BT: [N][K] bf16 (weights pre-transposed).
// 128x128 block tile; 256 threads = 8 waves, 2(M)x4(N); wave = 4x2 frags.
// Double-buffered async global->LDS staging (4 issues/wave/stage).
// mode 0: C[row*N+col] = acc (fp32)
// mode 1: Mamba split epilogue -> z (bf16), xbc (f32), dt (f32)

__global__ __launch_bounds__(256) void k_gemm_bf16(
    const bf16* __restrict__ A, const bf16* __restrict__ BT,
    int M, int N, int K,
    float* __restrict__ C, int mode,
    bf16* __restrict__ z_out, float* __restrict__ xbc_out,
    float* __restrict__ dt_out) {
  __shared__ bf16 sA[2][128][40];    // 128 rows x 32 k, +8 pad
  __shared__ bf16 sBt[2][128][40];   // 128 n-rows x 32 k, +8 pad

  int tid  = threadIdx.x;
  int wave = tid >> 5;
  int wm   = wave >> 2;              // 0..1
  int wn   = wave & 3;               // 0..3
  int tileM = blockIdx.y * 128;
  int tileN = blockIdx.x * 128;

  // per-thread fixed staging slots (fully unrolled: 4 async issues / thread)
  int sr  = tid >> 2;                // 0..63
  int sc  = (tid & 3) * 8;           // 0/8/16/24
  int nlast = N - 1 - tileN;
  int rc0 = (tileN + sr      < N) ? sr      : nlast;   // clamp OOB n-rows;
  int rc1 = (tileN + sr + 64 < N) ? sr + 64 : nlast;   // cols masked later

  v8f acc[4][2];
#pragma unroll
  for (int i = 0; i < 4; ++i)
#pragma unroll
    for (int j = 0; j < 2; ++j) acc[i][j] = zero8();

  auto stage = [&](int buf, int k0) {
    async_ld_b128(&sA[buf][sr][sc],      A + (size_t)(tileM + sr)      * K + k0 + sc);
    async_ld_b128(&sA[buf][sr + 64][sc], A + (size_t)(tileM + sr + 64) * K + k0 + sc);
    async_ld_b128(&sBt[buf][sr][sc],      BT + (size_t)(tileN + rc0) * K + k0 + sc);
    async_ld_b128(&sBt[buf][sr + 64][sc], BT + (size_t)(tileN + rc1) * K + k0 + sc);
  };

  const int nk = K >> 5;
  stage(0, 0);
  for (int kt = 0; kt < nk; ++kt) {
    int cur = kt & 1;
    if (kt + 1 < nk) {
      stage(cur ^ 1, (kt + 1) * 32);
      // speculative prefetch of the A stream two tiles ahead
      if (kt + 2 < nk)
        __builtin_prefetch(A + (size_t)(tileM + (tid >> 1)) * K + (kt + 2) * 32,
                           0, 1);
      WAIT_ASYNC(4);      // newest 4 still in flight; previous tile resident
    } else {
      WAIT_ASYNC(0);
    }
    __syncthreads();

    v16bf af[4];
#pragma unroll
    for (int i = 0; i < 4; ++i)
      af[i] = load_frag_a(&sA[cur][wm * 64 + i * 16][0], 40);
    v16bf bfr[2];
#pragma unroll
    for (int j = 0; j < 2; ++j)
      bfr[j] = load_frag_bt(&sBt[cur][wn * 32 + j * 16][0], 40);
#pragma unroll
    for (int i = 0; i < 4; ++i)
#pragma unroll
      for (int j = 0; j < 2; ++j)
        acc[i][j] = __builtin_amdgcn_wmma_f32_16x16x32_bf16(
            false, af[i], false, bfr[j], (short)0, acc[i][j], false, false);
    __syncthreads();
  }

  // epilogue (C/D layout: VGPR jj -> M=jj+8*(lane>=16), N=lane&15)
  int lane = tid & 31, nn = lane & 15, hh = lane >> 4;
#pragma unroll
  for (int i = 0; i < 4; ++i) {
#pragma unroll
    for (int j = 0; j < 2; ++j) {
      int r0 = tileM + wm * 64 + i * 16;
      int c0 = tileN + wn * 32 + j * 16;
#pragma unroll
      for (int jj = 0; jj < 8; ++jj) {
        int row = r0 + jj + 8 * hh;
        int col = c0 + nn;
        if (col >= N) continue;
        float val = acc[i][j][jj];
        if (mode == 0) {
          C[(size_t)row * N + col] = val;
        } else {
          if (col < D_INNER)
            z_out[(size_t)row * D_INNER + col] = (bf16)val;
          else if (col < D_INNER + CONV_DIM)
            xbc_out[(size_t)row * CONV_DIM + (col - D_INNER)] = val;
          else
            dt_out[(size_t)row * NHEADS + (col - D_INNER - CONV_DIM)] = val;
        }
      }
    }
  }
}

// ---------------- causal conv1d (K=4) + SiLU, fp32 -> bf16 ------------------

__global__ __launch_bounds__(256) void k_conv_silu(
    const float* __restrict__ xbc, const float* __restrict__ w,
    const float* __restrict__ bconv, bf16* __restrict__ out) {
  int ch  = blockIdx.x * 256 + threadIdx.x;   // < CONV_DIM (17*256)
  int b   = blockIdx.y >> 4;
  int seg = blockIdx.y & 15;
  int s0  = seg * 256;

  float w0 = w[ch * 4 + 0], w1 = w[ch * 4 + 1];
  float w2 = w[ch * 4 + 2], w3 = w[ch * 4 + 3];
  float bb = bconv[ch];

  size_t base = (size_t)b * SEQ;
  float x0 = (s0 - 3 >= 0) ? xbc[(base + s0 - 3) * CONV_DIM + ch] : 0.f;
  float x1 = (s0 - 2 >= 0) ? xbc[(base + s0 - 2) * CONV_DIM + ch] : 0.f;
  float x2 = (s0 - 1 >= 0) ? xbc[(base + s0 - 1) * CONV_DIM + ch] : 0.f;

  for (int s = s0; s < s0 + 256; ++s) {
    float x3 = xbc[(base + s) * CONV_DIM + ch];
    float acc = bb + w0 * x0 + w1 * x1 + w2 * x2 + w3 * x3;
    out[(base + s) * CONV_DIM + ch] = (bf16)siluf(acc);
    x0 = x1; x1 = x2; x2 = x3;
  }
}

// ---------------- dt softplus + per-chunk cumsum(dA) + chunk decay ----------

__global__ void k_dt(const float* __restrict__ dt_raw,
                     const float* __restrict__ dt_bias,
                     const float* __restrict__ Avec,
                     float* __restrict__ dt_proc,
                     float* __restrict__ dA_cs,
                     float* __restrict__ cdecay) {
  int t = blockIdx.x * 256 + threadIdx.x;     // < BATCH*NCHUNK*NHEADS = 4096
  int h = t & 63;
  int c = (t >> 6) & 31;
  int b = t >> 11;
  float bias = dt_bias[h];
  float a    = Avec[h];
  float cum  = 0.f;
  size_t rowbase = (size_t)b * SEQ + c * CHUNK;
  size_t dabase  = ((size_t)(b * NCHUNK + c) * NHEADS + h) * CHUNK;
  for (int i = 0; i < CHUNK; ++i) {
    float v  = dt_raw[(rowbase + i) * NHEADS + h] + bias;
    float sp = (v > 20.f) ? v : log1pf(__expf(v));
    dt_proc[(rowbase + i) * NHEADS + h] = sp;
    cum += sp * a;
    dA_cs[dabase + i] = cum;
  }
  cdecay[(size_t)(b * NCHUNK + c) * NHEADS + h] = __expf(cum);
}

// ---------------- SSD intra-chunk: Y_intra + chunk states (all WMMA) --------
// grid: (BATCH*NCHUNK, NHEADS); block 256 = 8 waves.

__global__ __launch_bounds__(256) void k_ssd_intra(
    const bf16* __restrict__ conv, const float* __restrict__ dt_proc,
    const float* __restrict__ dA_cs, float* __restrict__ y,
    bf16* __restrict__ states) {
  __shared__ bf16 sBm[CHUNK][D_STATE + 8];
  __shared__ bf16 sCm[CHUNK][D_STATE + 8];       // reused: M, then decay*Bm^T
  __shared__ bf16 sDtxT[HEAD_DIM][CHUNK + 8];    // dtx stored transposed [p][j]
  __shared__ float sDt[CHUNK], sDa[CHUNK], sDec[CHUNK];

  int bc = blockIdx.x;
  int h  = blockIdx.y;
  int tid = threadIdx.x;
  int b = bc / NCHUNK, c = bc % NCHUNK;
  size_t row0 = (size_t)b * SEQ + c * CHUNK;

  // fire async copies of Bm/Cm tiles first (16 issues / thread, unrolled)
#pragma unroll
  for (int vv = 0; vv < 8; ++vv) {
    int v = tid + vv * 256;
    int r = v >> 4, cv = (v & 15) * 8;
    const bf16* rowp = conv + (row0 + r) * CONV_DIM;
    async_ld_b128(&sBm[r][cv], rowp + D_INNER + cv);
    async_ld_b128(&sCm[r][cv], rowp + D_INNER + D_STATE + cv);
  }

  if (tid < CHUNK) {
    sDt[tid] = dt_proc[(row0 + tid) * NHEADS + h];
    sDa[tid] = dA_cs[((size_t)bc * NHEADS + h) * CHUNK + tid];
  }
  __syncthreads();
  if (tid < CHUNK) sDec[tid] = __expf(sDa[CHUNK - 1] - sDa[tid]);

  // dtx^T[p][j] = dt[j] * x[j][p]  (overlapped with async Bm/Cm copies)
#pragma unroll
  for (int vv = 0; vv < 4; ++vv) {
    int v = tid + vv * 256;
    int r = v >> 3, cv = (v & 7) * 8;
    union { uint4 u; bf16 e[8]; } t;
    t.u = *reinterpret_cast<const uint4*>(
              conv + (row0 + r) * CONV_DIM + (size_t)h * HEAD_DIM + cv);
    float dtv = sDt[r];
#pragma unroll
    for (int e = 0; e < 8; ++e)
      sDtxT[cv + e][r] = (bf16)((float)t.e[e] * dtv);
  }
  WAIT_ASYNC(0);
  __syncthreads();

  // ---- G = Cm @ Bm^T  (128x128, K=n=128)  waves 2x4 -----------------------
  int wave = tid >> 5;
  int wmG = wave >> 2, wnG = wave & 3;
  v8f g[4][2];
#pragma unroll
  for (int i = 0; i < 4; ++i)
#pragma unroll
    for (int j = 0; j < 2; ++j) g[i][j] = zero8();

  for (int k0 = 0; k0 < D_STATE; k0 += 32) {
    v16bf af[4], bfr[2];
#pragma unroll
    for (int i = 0; i < 4; ++i)
      af[i] = load_frag_a(&sCm[wmG * 64 + i * 16][k0], D_STATE + 8);
#pragma unroll
    for (int j = 0; j < 2; ++j)
      bfr[j] = load_frag_bt(&sBm[wnG * 32 + j * 16][k0], D_STATE + 8);
#pragma unroll
    for (int i = 0; i < 4; ++i)
#pragma unroll
      for (int j = 0; j < 2; ++j)
        g[i][j] = __builtin_amdgcn_wmma_f32_16x16x32_bf16(
            false, af[i], false, bfr[j], (short)0, g[i][j], false, false);
  }
  __syncthreads();   // everyone done reading sCm

  // ---- M = G * L  (causal exp mask), overwrite sCm ------------------------
  {
    int lane = tid & 31, nn = lane & 15, hh = lane >> 4;
#pragma unroll
    for (int i = 0; i < 4; ++i) {
#pragma unroll
      for (int j = 0; j < 2; ++j) {
        int r0 = wmG * 64 + i * 16;
        int c0 = wnG * 32 + j * 16;
#pragma unroll
        for (int jj = 0; jj < 8; ++jj) {
          int ri = r0 + jj + 8 * hh;
          int cj = c0 + nn;
          float val =
              (cj <= ri) ? g[i][j][jj] * __expf(sDa[ri] - sDa[cj]) : 0.f;
          sCm[ri][cj] = (bf16)val;
        }
      }
    }
  }
  __syncthreads();

  // ---- Y_intra = M @ dtx  (128x64, K=j=128)  waves 4x2 --------------------
  int wmY = wave >> 1, wnY = wave & 1;
  v8f yacc[2][2];
#pragma unroll
  for (int i = 0; i < 2; ++i)
#pragma unroll
    for (int j = 0; j < 2; ++j) yacc[i][j] = zero8();

  for (int k0 = 0; k0 < CHUNK; k0 += 32) {
    v16bf af[2], bfr[2];
#pragma unroll
    for (int i = 0; i < 2; ++i)
      af[i] = load_frag_a(&sCm[wmY * 32 + i * 16][k0], D_STATE + 8);
#pragma unroll
    for (int j = 0; j < 2; ++j)
      bfr[j] = load_frag_bt(&sDtxT[wnY * 32 + j * 16][k0], CHUNK + 8);
#pragma unroll
    for (int i = 0; i < 2; ++i)
#pragma unroll
      for (int j = 0; j < 2; ++j)
        yacc[i][j] = __builtin_amdgcn_wmma_f32_16x16x32_bf16(
            false, af[i], false, bfr[j], (short)0, yacc[i][j], false, false);
  }

  int lane = tid & 31, nn = lane & 15, hh = lane >> 4;
#pragma unroll
  for (int i = 0; i < 2; ++i) {
#pragma unroll
    for (int j = 0; j < 2; ++j) {
      int r0 = wmY * 32 + i * 16;
      int c0 = wnY * 32 + j * 16;
#pragma unroll
      for (int jj = 0; jj < 8; ++jj) {
        int row = r0 + jj + 8 * hh;
        int col = c0 + nn;
        y[(row0 + row) * D_INNER + (size_t)h * HEAD_DIM + col] = yacc[i][j][jj];
      }
    }
  }
  __syncthreads();   // all waves done reading sCm (M)

  // ---- rebuild sCm as (decay ∘ Bm)^T : [n][j] -----------------------------
  for (int vv = 0; vv < 64; ++vv) {
    int v = tid + vv * 256;
    int n_ = v >> 7, j_ = v & 127;
    sCm[n_][j_] = (bf16)((float)sBm[j_][n_] * sDec[j_]);
  }
  __syncthreads();

  // ---- states = (Bm ∘ decay)^T @ dtx  (128x64 over n x p, K=j=128) --------
  v8f st[2][2];
#pragma unroll
  for (int i = 0; i < 2; ++i)
#pragma unroll
    for (int j = 0; j < 2; ++j) st[i][j] = zero8();

  for (int k0 = 0; k0 < CHUNK; k0 += 32) {
    v16bf af[2], bfr[2];
#pragma unroll
    for (int i = 0; i < 2; ++i)
      af[i] = load_frag_a(&sCm[wmY * 32 + i * 16][k0], D_STATE + 8);
#pragma unroll
    for (int j = 0; j < 2; ++j)
      bfr[j] = load_frag_bt(&sDtxT[wnY * 32 + j * 16][k0], CHUNK + 8);
#pragma unroll
    for (int i = 0; i < 2; ++i)
#pragma unroll
      for (int j = 0; j < 2; ++j)
        st[i][j] = __builtin_amdgcn_wmma_f32_16x16x32_bf16(
            false, af[i], false, bfr[j], (short)0, st[i][j], false, false);
  }

  size_t sbase = ((size_t)bc * NHEADS + h) * D_STATE * HEAD_DIM;
#pragma unroll
  for (int i = 0; i < 2; ++i) {
#pragma unroll
    for (int j = 0; j < 2; ++j) {
      int n0 = wmY * 32 + i * 16;
      int p0 = wnY * 32 + j * 16;
#pragma unroll
      for (int jj = 0; jj < 8; ++jj) {
        int n_ = n0 + jj + 8 * hh;
        int p_ = p0 + nn;
        states[sbase + (size_t)n_ * HEAD_DIM + p_] = (bf16)st[i][j][jj];
      }
    }
  }
}

// ---------------- inter-chunk recurrence scan (in-place states -> prevs) ---

__global__ void k_scan(bf16* __restrict__ states,
                       const float* __restrict__ cdecay) {
  int t = blockIdx.x * 256 + threadIdx.x;     // < BATCH*NHEADS*D_STATE*HEAD_DIM
  int p = t & 63;
  int n = (t >> 6) & 127;
  int h = (t >> 13) & 63;
  int b = (t >> 19) & 1;
  float running = 0.f;
  for (int c = 0; c < NCHUNK; ++c) {
    size_t idx = (((size_t)(b * NCHUNK + c) * NHEADS + h) * D_STATE + n) *
                     HEAD_DIM + p;
    float s = (float)states[idx];
    states[idx] = (bf16)running;                        // prev state for chunk c
    running = cdecay[(size_t)(b * NCHUNK + c) * NHEADS + h] * running + s;
  }
}

// ---------------- SSD inter-chunk: Y += (C ∘ exp(dA_cs)) @ prev + D*x ------

__global__ __launch_bounds__(256) void k_ssd_inter(
    const bf16* __restrict__ conv, const float* __restrict__ dA_cs,
    const bf16* __restrict__ prevs, const float* __restrict__ Dvec,
    float* __restrict__ y) {
  __shared__ bf16 sCs[CHUNK][D_STATE + 8];
  __shared__ bf16 sPrevT[HEAD_DIM][D_STATE + 8];   // prev stored transposed [p][n]
#if HAVE_TDM
  __shared__ bf16 sPrevRaw[D_STATE][HEAD_DIM];     // TDM landing pad [n][p]
#endif
  __shared__ float sDa[CHUNK];

  int bc = blockIdx.x;
  int h  = blockIdx.y;
  int tid = threadIdx.x;
  int b = bc / NCHUNK, c = bc % NCHUNK;
  size_t row0 = (size_t)b * SEQ + c * CHUNK;
  size_t sbase = ((size_t)bc * NHEADS + h) * D_STATE * HEAD_DIM;

#if HAVE_TDM
  // One TDM descriptor per workgroup (wave 0 issues): 2D tile, 128x64 bf16,
  // contiguous in global -> sPrevRaw. Tracked by TENSORcnt.
  if (tid < 32) {
    unsigned long long ga =
        (unsigned long long)(const void*)(prevs + sbase);
    unsigned lds_base = (unsigned)(unsigned long long)(
        (__attribute__((address_space(3))) char*)&sPrevRaw[0][0]);
    v4u g0 = {
        1u,                                          // count=1, user D#
        lds_base,                                    // lds_addr
        (unsigned)(ga & 0xFFFFFFFFu),                // global_addr[31:0]
        (unsigned)((ga >> 32) & 0x01FFFFFFu) | (2u << 30)  // ga[56:32]|type=2
    };
    v8i g1 = {
        0x00010000,                                  // data_size=2B
        (int)(HEAD_DIM << 16),                       // tensor_dim0 = 64
        (int)(D_STATE << 16),                        // tensor_dim1 = 128
        (int)(HEAD_DIM << 16),                       // tile_dim0 = 64
        (int)D_STATE,                                // tile_dim1 = 128
        (int)HEAD_DIM,                               // tensor_dim0_stride = 64
        0, 0
    };
    v4i gz = {0, 0, 0, 0};
    v8i gz8 = {0, 0, 0, 0, 0, 0, 0, 0};
    __builtin_amdgcn_tensor_load_to_lds(g0, g1, gz, gz, gz8, 0);
  }
#endif

  if (tid < CHUNK)
    sDa[tid] = dA_cs[((size_t)bc * NHEADS + h) * CHUNK + tid];
  __syncthreads();

  // sCs = Cm scaled by exp(dA_cs[i]) per row (overlaps the TDM transfer)
#pragma unroll
  for (int vv = 0; vv < 8; ++vv) {
    int v = tid + vv * 256;
    int r = v >> 4, cv = (v & 15) * 8;
    union { uint4 u; bf16 e[8]; } t;
    t.u = *reinterpret_cast<const uint4*>(
              conv + (row0 + r) * CONV_DIM + D_INNER + D_STATE + cv);
    float sc = __expf(sDa[r]);
#pragma unroll
    for (int e = 0; e < 8; ++e)
      sCs[r][cv + e] = (bf16)((float)t.e[e] * sc);
  }

#if HAVE_TDM
  if (tid < 32) __builtin_amdgcn_s_wait_tensorcnt(0);
  __syncthreads();
  // LDS->LDS transpose into fragment-friendly layout sPrevT[p][n]
#pragma unroll
  for (int vv = 0; vv < 4; ++vv) {
    int v = tid + vv * 256;
    int r = v >> 3, cv = (v & 7) * 8;
#pragma unroll
    for (int e = 0; e < 8; ++e) sPrevT[cv + e][r] = sPrevRaw[r][cv + e];
  }
#else
  // direct global staging of sPrevT[p][n] from states [n][p]
#pragma unroll
  for (int vv = 0; vv < 4; ++vv) {
    int v = tid + vv * 256;
    int r = v >> 3, cv = (v & 7) * 8;
    union { uint4 u; bf16 e[8]; } t;
    t.u = *reinterpret_cast<const uint4*>(prevs + sbase +
                                          (size_t)r * HEAD_DIM + cv);
#pragma unroll
    for (int e = 0; e < 8; ++e) sPrevT[cv + e][r] = t.e[e];
  }
#endif
  __syncthreads();

  int wave = tid >> 5;
  int wm = wave >> 1, wn = wave & 1;          // 4x2 waves over 128x64
  v8f acc[2][2];
#pragma unroll
  for (int i = 0; i < 2; ++i)
#pragma unroll
    for (int j = 0; j < 2; ++j) acc[i][j] = zero8();

  for (int k0 = 0; k0 < D_STATE; k0 += 32) {
    v16bf af[2], bfr[2];
#pragma unroll
    for (int i = 0; i < 2; ++i)
      af[i] = load_frag_a(&sCs[wm * 32 + i * 16][k0], D_STATE + 8);
#pragma unroll
    for (int j = 0; j < 2; ++j)
      bfr[j] = load_frag_bt(&sPrevT[wn * 32 + j * 16][k0], D_STATE + 8);
#pragma unroll
    for (int i = 0; i < 2; ++i)
#pragma unroll
      for (int j = 0; j < 2; ++j)
        acc[i][j] = __builtin_amdgcn_wmma_f32_16x16x32_bf16(
            false, af[i], false, bfr[j], (short)0, acc[i][j], false, false);
  }

  float Dh = Dvec[h];
  int lane = tid & 31, nn = lane & 15, hh = lane >> 4;
#pragma unroll
  for (int i = 0; i < 2; ++i) {
#pragma unroll
    for (int j = 0; j < 2; ++j) {
      int r0 = wm * 32 + i * 16;
      int c0 = wn * 32 + j * 16;
#pragma unroll
      for (int jj = 0; jj < 8; ++jj) {
        int row = r0 + jj + 8 * hh;
        int col = c0 + nn;
        float xv = (float)conv[(row0 + row) * CONV_DIM + (size_t)h * HEAD_DIM + col];
        size_t idx = (row0 + row) * D_INNER + (size_t)h * HEAD_DIM + col;
        y[idx] += acc[i][j][jj] + Dh * xv;
      }
    }
  }
}

// ---------------- gating (silu(z)) + RMSNorm -> bf16 ------------------------

__global__ __launch_bounds__(256) void k_gate_norm(
    const float* __restrict__ y, const bf16* __restrict__ z,
    const float* __restrict__ norm_w, bf16* __restrict__ yn) {
  __shared__ float red[256];
  int row = blockIdx.x;
  int tid = threadIdx.x;
  float v[16];
  float ss = 0.f;
#pragma unroll
  for (int k = 0; k < 16; ++k) {
    int col = tid + k * 256;
    float zv = (float)z[(size_t)row * D_INNER + col];
    float val = y[(size_t)row * D_INNER + col] * siluf(zv);
    v[k] = val;
    ss += val * val;
  }
  red[tid] = ss;
  __syncthreads();
  for (int s = 128; s > 0; s >>= 1) {
    if (tid < s) red[tid] += red[tid + s];
    __syncthreads();
  }
  float rms = rsqrtf(red[0] / (float)D_INNER + RMS_EPS);
#pragma unroll
  for (int k = 0; k < 16; ++k) {
    int col = tid + k * 256;
    yn[(size_t)row * D_INNER + col] = (bf16)(v[k] * rms * norm_w[col]);
  }
}

// ---------------------------------------------------------------------------

extern "C" void kernel_launch(void* const* d_in, const int* in_sizes, int n_in,
                              void* d_out, int out_size, void* d_ws,
                              size_t ws_size, hipStream_t stream) {
  (void)in_sizes; (void)n_in; (void)out_size; (void)ws_size;

  const float* hs      = (const float*)d_in[0];
  const float* W_in    = (const float*)d_in[1];
  const float* conv_w  = (const float*)d_in[2];
  const float* conv_b  = (const float*)d_in[3];
  const float* Avec    = (const float*)d_in[4];
  const float* Dvec    = (const float*)d_in[5];
  const float* dt_bias = (const float*)d_in[6];
  const float* norm_w  = (const float*)d_in[7];
  const float* W_out   = (const float*)d_in[8];
  float* out = (float*)d_out;

  // ---- workspace carving (disjoint-lifetime aliasing) ----
  char* ws = (char*)d_ws;
  size_t off = 0;
  auto take = [&](size_t bytes) -> char* {
    char* p = ws + off;
    off += (bytes + 255) & ~(size_t)255;
    return p;
  };
  bf16*  hs_bf    = (bf16*) take((size_t)BATCH * SEQ * D_MODEL * 2);
  bf16*  WinT_bf  = (bf16*) take((size_t)D_MODEL * D_IN_PROJ * 2);  // [N][K]
  bf16*  WoutT_bf = (bf16*) take((size_t)D_INNER * D_MODEL * 2);    // [N][K]
  bf16*  z_bf     = (bf16*) take((size_t)BATCH * SEQ * D_INNER * 2);
  float* xbc_f    = (float*)take((size_t)BATCH * SEQ * CONV_DIM * 4);
  float* y_f      = xbc_f;                        // alias: xbc dead after conv
  float* dt_raw   = (float*)take((size_t)BATCH * SEQ * NHEADS * 4);
  float* dt_pr    = (float*)take((size_t)BATCH * SEQ * NHEADS * 4);
  float* dA_cs    = (float*)take((size_t)BATCH * NCHUNK * NHEADS * CHUNK * 4);
  float* cdec     = (float*)take((size_t)BATCH * NCHUNK * NHEADS * 4);
  bf16*  convout  = (bf16*) take((size_t)BATCH * SEQ * CONV_DIM * 2);
  bf16*  yn_bf    = convout;                      // alias: conv dead after inter
  bf16*  states   = (bf16*) take((size_t)BATCH * NCHUNK * NHEADS * D_STATE *
                                 HEAD_DIM * 2);

  // 1) fp32 -> bf16 conversions (weights transposed to [N][K])
  {
    int n1 = BATCH * SEQ * D_MODEL;
    k_f32_to_bf16<<<n1 / 1024, 256, 0, stream>>>(hs, hs_bf, n1);
    k_f32_to_bf16_T<<<dim3(D_IN_PROJ / 32, D_MODEL / 32), 256, 0, stream>>>(
        W_in, WinT_bf, D_MODEL, D_IN_PROJ);
    k_f32_to_bf16_T<<<dim3(D_MODEL / 32, D_INNER / 32), 256, 0, stream>>>(
        W_out, WoutT_bf, D_INNER, D_MODEL);
  }

  // 2) in_proj GEMM with split epilogue: z | xbc | dt
  {
    dim3 grid((D_IN_PROJ + 127) / 128, (BATCH * SEQ) / 128);
    k_gemm_bf16<<<grid, 256, 0, stream>>>(hs_bf, WinT_bf, BATCH * SEQ,
                                          D_IN_PROJ, D_MODEL, nullptr, 1,
                                          z_bf, xbc_f, dt_raw);
  }

  // 3) causal conv1d + SiLU  -> bf16 [x|B|C]
  k_conv_silu<<<dim3(CONV_DIM / 256, BATCH * 16), 256, 0, stream>>>(
      xbc_f, conv_w, conv_b, convout);

  // 4) dt softplus, dA cumsum per chunk, chunk decay
  k_dt<<<(BATCH * NCHUNK * NHEADS) / 256, 256, 0, stream>>>(
      dt_raw, dt_bias, Avec, dt_pr, dA_cs, cdec);

  // 5) intra-chunk WMMA: Y_intra + chunk states
  k_ssd_intra<<<dim3(BATCH * NCHUNK, NHEADS), 256, 0, stream>>>(
      convout, dt_pr, dA_cs, y_f, states);

  // 6) inter-chunk recurrence scan (in-place states -> prev-states)
  k_scan<<<(BATCH * NHEADS * D_STATE * HEAD_DIM) / 256, 256, 0, stream>>>(
      states, cdec);

  // 7) inter-chunk WMMA: Y += (C ∘ exp(dA_cs)) @ prev + D*x
  k_ssd_inter<<<dim3(BATCH * NCHUNK, NHEADS), 256, 0, stream>>>(
      convout, dA_cs, states, Dvec, y_f);

  // 8) gating + RMSNorm -> bf16
  k_gate_norm<<<BATCH * SEQ, 256, 0, stream>>>(y_f, z_bf, norm_w, yn_bf);

  // 9) out_proj GEMM -> fp32 output
  {
    dim3 grid(D_MODEL / 128, (BATCH * SEQ) / 128);
    k_gemm_bf16<<<grid, 256, 0, stream>>>(yn_bf, WoutT_bf, BATCH * SEQ,
                                          D_MODEL, D_INNER, out, 0,
                                          nullptr, nullptr, nullptr);
  }
}
